// GAT_65515431133472
// MI455X (gfx1250) — matrix operand order repaired and checked by reading, not compiled, not run on hardware
//
#include <hip/hip_runtime.h>
#include <math.h>

#define F_IN   256
#define HC     128   // H*C
#define NHEAD  4
#define CPH    32
#define NEG_SLOPE 0.2f

typedef __attribute__((ext_vector_type(16))) __bf16 v16bf;
typedef __attribute__((ext_vector_type(8)))  float  v8f;

union Frag16 {
  v16bf v;
  unsigned short u[16];
  uint4 q[2];
};

static __device__ __forceinline__ unsigned short f2bf(float f) {
  unsigned u = __float_as_uint(f);
  u += 0x7fffu + ((u >> 16) & 1u);   // round-to-nearest-even
  return (unsigned short)(u >> 16);
}

// Order-preserving float<->int encoding (involution) so atomicMax(int) == float max.
static __device__ __forceinline__ int fflip(int i) { return i ^ ((i >> 31) & 0x7fffffff); }

// ---------------- init: out = bias broadcast; m = -inf (encoded); s = 0 ----------------
__global__ void k_init(float* __restrict__ out, const float* __restrict__ bias,
                       int* __restrict__ m_enc, float* __restrict__ s, int N) {
  int idx = blockIdx.x * blockDim.x + threadIdx.x;
  if (idx < N * HC) out[idx] = bias[idx & (HC - 1)];
  if (idx < N * NHEAD) { m_enc[idx] = fflip(__float_as_int(-INFINITY)); s[idx] = 0.f; }
}

// ---------------- f32 -> bf16 conversions ----------------
__global__ void k_cvt_x(const float* __restrict__ x, unsigned short* __restrict__ xb, int total) {
  int idx = blockIdx.x * blockDim.x + threadIdx.x;
  if (idx < total) xb[idx] = f2bf(x[idx]);
}

__global__ void k_cvt_w(const float* __restrict__ W, unsigned short* __restrict__ Wt) {
  int idx = blockIdx.x * blockDim.x + threadIdx.x;   // HC * F_IN
  if (idx < HC * F_IN) {
    int n = idx / F_IN, k = idx % F_IN;
    Wt[idx] = f2bf(W[k * HC + n]);                   // Wt[n][k]: per-column contiguous K
  }
}

// ---------------- WMMA GEMM: h[N,128] = x @ W (bf16 in, f32 acc) ----------------
// One wave computes a 16x128 row-block: 8 column tiles of 16x16, K-loop over 256 in steps of 32.
// All 8 B fragments for a k-step are preloaded before the WMMA chain so the 16 b128 loads
// can be in flight simultaneously (MLP) instead of load->wait->wmma serialization.
__global__ void __launch_bounds__(256) k_gemm(const unsigned short* __restrict__ xb,
                                              const unsigned short* __restrict__ Wt,
                                              float* __restrict__ h, int N) {
  int lane = threadIdx.x & 31;
  int wave = threadIdx.x >> 5;
  int rowBase = (blockIdx.x * 8 + wave) * 16;
  if (rowBase >= N) return;                 // wave-uniform: EXEC stays all-ones for WMMA
  int m  = lane & 15;
  int hi = lane >> 4;
  int row = rowBase + m; if (row > N - 1) row = N - 1;   // clamp tail rows (stores guarded)
  const unsigned short* xrow = xb + (size_t)row * F_IN;
  int n = m;

  v8f acc[8];
  v8f zero = {};
  #pragma unroll
  for (int t = 0; t < 8; ++t) acc[t] = zero;

  for (int k0 = 0; k0 < F_IN; k0 += 32) {
    // A fragment (16x32 bf16): lanes 0-15 -> K {0..7,16..23}, lanes 16-31 -> K {8..15,24..31}
    Frag16 a;
    a.q[0] = *(const uint4*)(xrow + k0 + hi * 8);
    a.q[1] = *(const uint4*)(xrow + k0 + 16 + hi * 8);

    // Preload all 8 B fragments (32x16 bf16 each): lane column n,
    // lanes 0-15 -> K {0..7,16..23}, lanes 16-31 -> K {8..15,24..31}
    Frag16 b[8];
    #pragma unroll
    for (int nt = 0; nt < 8; ++nt) {
      const unsigned short* wcol = Wt + (size_t)(nt * 16 + n) * F_IN + k0 + hi * 8;
      b[nt].q[0] = *(const uint4*)(wcol);
      b[nt].q[1] = *(const uint4*)(wcol + 16);
    }

    #pragma unroll
    for (int nt = 0; nt < 8; ++nt) {
      acc[nt] = __builtin_amdgcn_wmma_f32_16x16x32_bf16(false, a.v, false, b[nt].v,
                                                        (short)0, acc[nt], false, false);
    }
  }

  // D layout: VGPR r holds M = r + 8*hi, N = lane&15
  #pragma unroll
  for (int nt = 0; nt < 8; ++nt) {
    #pragma unroll
    for (int r = 0; r < 8; ++r) {
      int rr = rowBase + r + hi * 8;
      if (rr < N) h[(size_t)rr * HC + nt * 16 + n] = acc[nt][r];
    }
  }
}

// ---------------- per-node attention halves ----------------
__global__ void k_att(const float* __restrict__ h, const float* __restrict__ att_src,
                      const float* __restrict__ att_dst, float* __restrict__ a_src,
                      float* __restrict__ a_dst, int N) {
  int idx = blockIdx.x * blockDim.x + threadIdx.x;
  if (idx >= N * NHEAD) return;
  int node = idx >> 2, head = idx & 3;
  const float* hp = h + (size_t)node * HC + head * CPH;
  const float* as = att_src + head * CPH;
  const float* ad = att_dst + head * CPH;
  float s1 = 0.f, s2 = 0.f;
  #pragma unroll 8
  for (int c = 0; c < CPH; ++c) { float v = hp[c]; s1 += v * as[c]; s2 += v * ad[c]; }
  a_src[idx] = s1; a_dst[idx] = s2;
}

static __device__ __forceinline__ void resolve_edge(const int* __restrict__ ei, int idx, int E,
                                                    int& sN, int& dN) {
  if (idx < E) { sN = ei[idx]; dN = ei[E + idx]; }
  else         { sN = dN = idx - E; }          // self-loops appended after real edges
}

// ---------------- pass 1: segment max of leaky-relu logits ----------------
__global__ void k_edge_max(const int* __restrict__ ei, const float* __restrict__ a_src,
                           const float* __restrict__ a_dst, int* __restrict__ m_enc,
                           int E, int Etot) {
  int idx = blockIdx.x * blockDim.x + threadIdx.x;
  if (idx >= Etot) return;
  int sN, dN; resolve_edge(ei, idx, E, sN, dN);
  float4 as = *(const float4*)(a_src + (size_t)sN * 4);
  float4 ad = *(const float4*)(a_dst + (size_t)dN * 4);
  float ev[4] = {as.x + ad.x, as.y + ad.y, as.z + ad.z, as.w + ad.w};
  #pragma unroll
  for (int t = 0; t < 4; ++t) {
    float e = ev[t] > 0.f ? ev[t] : ev[t] * NEG_SLOPE;
    atomicMax(&m_enc[dN * 4 + t], fflip(__float_as_int(e)));
  }
}

// ---------------- pass 2: segment sum of exp(e - m) ----------------
__global__ void k_edge_sum(const int* __restrict__ ei, const float* __restrict__ a_src,
                           const float* __restrict__ a_dst, const int* __restrict__ m_enc,
                           float* __restrict__ s, int E, int Etot) {
  int idx = blockIdx.x * blockDim.x + threadIdx.x;
  if (idx >= Etot) return;
  int sN, dN; resolve_edge(ei, idx, E, sN, dN);
  float4 as = *(const float4*)(a_src + (size_t)sN * 4);
  float4 ad = *(const float4*)(a_dst + (size_t)dN * 4);
  float ev[4] = {as.x + ad.x, as.y + ad.y, as.z + ad.z, as.w + ad.w};
  #pragma unroll
  for (int t = 0; t < 4; ++t) {
    float e = ev[t] > 0.f ? ev[t] : ev[t] * NEG_SLOPE;
    float m = __int_as_float(fflip(m_enc[dN * 4 + t]));
    atomicAdd(&s[dN * 4 + t], __expf(e - m));
  }
}

// ---------------- pass 3: weighted scatter-aggregate, one wave per edge ----------------
__global__ void __launch_bounds__(256) k_scatter(const int* __restrict__ ei,
                                                 const float* __restrict__ a_src,
                                                 const float* __restrict__ a_dst,
                                                 const int* __restrict__ m_enc,
                                                 const float* __restrict__ s,
                                                 const float* __restrict__ h,
                                                 float* __restrict__ out, int E, int Etot) {
  int edge = blockIdx.x * 8 + (threadIdx.x >> 5);
  int lane = threadIdx.x & 31;
  if (edge >= Etot) return;
  int sN, dN; resolve_edge(ei, edge, E, sN, dN);
  int head = lane >> 3;                         // 8 lanes per head (4 channels each)
  float av = a_src[sN * 4 + head] + a_dst[dN * 4 + head];
  float e  = av > 0.f ? av : av * NEG_SLOPE;
  float m  = __int_as_float(fflip(m_enc[dN * 4 + head]));
  float alpha = __expf(e - m) / (s[dN * 4 + head] + 1e-16f);
  float4 hv = *(const float4*)(h + (size_t)sN * HC + lane * 4);
  float* op = out + (size_t)dN * HC + lane * 4;
  atomicAdd(op + 0, hv.x * alpha);
  atomicAdd(op + 1, hv.y * alpha);
  atomicAdd(op + 2, hv.z * alpha);
  atomicAdd(op + 3, hv.w * alpha);
}

extern "C" void kernel_launch(void* const* d_in, const int* in_sizes, int n_in,
                              void* d_out, int out_size, void* d_ws, size_t ws_size,
                              hipStream_t stream) {
  const float* x       = (const float*)d_in[0];
  const int*   ei      = (const int*)  d_in[1];
  const float* W       = (const float*)d_in[2];
  const float* att_src = (const float*)d_in[3];
  const float* att_dst = (const float*)d_in[4];
  const float* bias    = (const float*)d_in[5];
  int N    = in_sizes[0] / F_IN;
  int E    = in_sizes[1] / 2;
  int Etot = E + N;
  float* out = (float*)d_out;

  char* ws = (char*)d_ws;
  size_t off = 0;
  auto wsalloc = [&](size_t bytes) -> char* {
    char* p = ws + off; off += (bytes + 255) & ~(size_t)255; return p;
  };
  float*          h     = (float*)         wsalloc((size_t)N * HC * sizeof(float));
  unsigned short* xb    = (unsigned short*)wsalloc((size_t)N * F_IN * sizeof(unsigned short));
  unsigned short* Wt    = (unsigned short*)wsalloc((size_t)HC * F_IN * sizeof(unsigned short));
  float*          a_src = (float*)         wsalloc((size_t)N * NHEAD * sizeof(float));
  float*          a_dst = (float*)         wsalloc((size_t)N * NHEAD * sizeof(float));
  int*            m_enc = (int*)           wsalloc((size_t)N * NHEAD * sizeof(int));
  float*          sbuf  = (float*)         wsalloc((size_t)N * NHEAD * sizeof(float));
  (void)ws_size; (void)n_in; (void)out_size;

  const int blk = 256;
  int xtotal = N * F_IN;
  k_init   <<<(N * HC + blk - 1) / blk, blk, 0, stream>>>(out, bias, m_enc, sbuf, N);
  k_cvt_x  <<<(xtotal + blk - 1) / blk, blk, 0, stream>>>(x, xb, xtotal);
  k_cvt_w  <<<(HC * F_IN + blk - 1) / blk, blk, 0, stream>>>(W, Wt);
  k_gemm   <<<(N + 127) / 128, 256, 0, stream>>>(xb, Wt, h, N);
  k_att    <<<(N * NHEAD + blk - 1) / blk, blk, 0, stream>>>(h, att_src, att_dst, a_src, a_dst, N);
  k_edge_max<<<(Etot + blk - 1) / blk, blk, 0, stream>>>(ei, a_src, a_dst, m_enc, E, Etot);
  k_edge_sum<<<(Etot + blk - 1) / blk, blk, 0, stream>>>(ei, a_src, a_dst, m_enc, sbuf, E, Etot);
  k_scatter<<<(Etot + 7) / 8, 256, 0, stream>>>(ei, a_src, a_dst, m_enc, sbuf, h, out, E, Etot);
}
